// CausalSelfAttention_32530082300548
// MI455X (gfx1250) — matrix operand order repaired and checked by reading, not compiled, not run on hardware
//
#include <hip/hip_runtime.h>
#include <hip/hip_bf16.h>

// ---------------------------------------------------------------------------
// CDNA5 (gfx1250) causal self-attention: bf16 WMMA pipeline
//   x[B,T,C] f32 -> qkv proj -> flash attention -> out proj -> f32
//   B=4 T=2048 C=1024 H=16 D=64
// ---------------------------------------------------------------------------

typedef __attribute__((ext_vector_type(16))) __bf16 v16bf;
typedef __attribute__((ext_vector_type(8)))  float  v8f;
typedef __attribute__((ext_vector_type(4)))  unsigned int u32x4;

union FragB16 {
    u32x4 u[2];
    v16bf v;
};

__device__ __forceinline__ unsigned short f32_to_bf16(float f) {
    unsigned int u = __float_as_uint(f);
    unsigned int r = u + 0x7FFFu + ((u >> 16) & 1u);   // round-to-nearest-even
    return (unsigned short)(r >> 16);
}

// ---------------------------------------------------------------------------
// Stage 0a: fp32 -> bf16 conversion (activations)
// ---------------------------------------------------------------------------
__global__ void cvt_f32_to_bf16(const float* __restrict__ in,
                                unsigned short* __restrict__ out, int n) {
    int i = blockIdx.x * blockDim.x + threadIdx.x;
    int stride = gridDim.x * blockDim.x;
    for (; i < n; i += stride) out[i] = f32_to_bf16(in[i]);
}

// ---------------------------------------------------------------------------
// Stage 0b: fp32 [K,N] -> bf16 [N,K] transpose-convert (weights, one-time)
// ---------------------------------------------------------------------------
__global__ void cvt_transpose_f32_to_bf16(const float* __restrict__ in,
                                          unsigned short* __restrict__ out,
                                          int K, int N) {
    int idx = blockIdx.x * blockDim.x + threadIdx.x;
    int total = K * N;
    int stride = gridDim.x * blockDim.x;
    for (; idx < total; idx += stride) {
        int n = idx / K, k = idx - n * K;          // out[n][k] = in[k][n]
        out[idx] = f32_to_bf16(in[(size_t)k * N + n]);
    }
}

// ---------------------------------------------------------------------------
// GEMM tiling: block = 256 threads = 8 waves; tile = 128(M) x 128(N); BK = 32
//   waves 4(M) x 2(N); each wave computes 32x64 (2 A-frags x 4 B-frags ->
//   8 WMMAs per K-step). LDS double-buffered: one barrier per K-step,
//   next tile's global->LDS fill overlaps current tile's WMMA burst.
// ---------------------------------------------------------------------------
#define BM   128
#define BN   128
#define BK   32
#define LDT  40   // bf16 pitch (32 + 8 pad) -> 80B rows, 16B aligned

// ---------------------------------------------------------------------------
// Stage 1: QKV projection. A = xb [8192,1024] bf16, Bt = W_in^T bf16 [3072,1024]
// Epilogue: +bias, scatter q,k -> [BH,T,64], v -> transposed [BH,64,T]
// ---------------------------------------------------------------------------
__global__ __launch_bounds__(256) void gemm_qkv_bf16(
    const unsigned short* __restrict__ A,
    const unsigned short* __restrict__ Bt,   // [N,K] pre-transposed
    const float* __restrict__ bias,
    unsigned short* __restrict__ qb,
    unsigned short* __restrict__ kb,
    unsigned short* __restrict__ vt)
{
    const int K = 1024, T = 2048;
    __shared__ __align__(16) unsigned short lA[2][BM * LDT];
    __shared__ __align__(16) unsigned short lB[2][BN * LDT];

    const int tid  = threadIdx.x;
    const int wave = tid >> 5;
    const int lane = tid & 31;
    const int h    = lane >> 4;
    const int ln   = lane & 15;
    const int wm   = wave & 3;       // 4 waves over M (32 rows each)
    const int wn   = wave >> 2;      // 2 waves over N (64 cols each)
    const int rowBase = blockIdx.y * BM;
    const int colBase = blockIdx.x * BN;

    v8f acc[2][4];
#pragma unroll
    for (int i = 0; i < 2; ++i)
#pragma unroll
        for (int j = 0; j < 4; ++j) acc[i][j] = (v8f)0.0f;

    const int fr = tid >> 2;             // fill row 0..63 (x2 halves)
    const int fc = (tid & 3) * 8;        // fill col chunk

    // prologue: fill buffer 0 with K-step 0
#pragma unroll
    for (int it = 0; it < 2; ++it) {
        int r = fr + it * 64;
        *(u32x4*)(&lA[0][r * LDT + fc]) =
            *(const u32x4*)(A + (size_t)(rowBase + r) * K + fc);
        *(u32x4*)(&lB[0][r * LDT + fc]) =
            *(const u32x4*)(Bt + (size_t)(colBase + r) * K + fc);
    }
    __syncthreads();

    for (int k0 = 0; k0 < K; k0 += BK) {
        const int cur = (k0 >> 5) & 1;
        const int nxt = cur ^ 1;

        // fill next buffer (overlaps with this step's WMMAs)
        if (k0 + BK < K) {
            if (k0 + 2 * BK < K) {   // prefetch two steps ahead
                __builtin_prefetch(A  + (size_t)(rowBase + fr) * K + k0 + 2 * BK + fc, 0, 3);
                __builtin_prefetch(Bt + (size_t)(colBase + fr) * K + k0 + 2 * BK + fc, 0, 3);
            }
#pragma unroll
            for (int it = 0; it < 2; ++it) {
                int r = fr + it * 64;
                *(u32x4*)(&lA[nxt][r * LDT + fc]) =
                    *(const u32x4*)(A + (size_t)(rowBase + r) * K + k0 + BK + fc);
                *(u32x4*)(&lB[nxt][r * LDT + fc]) =
                    *(const u32x4*)(Bt + (size_t)(colBase + r) * K + k0 + BK + fc);
            }
        }

        // preload fragments from current buffer, burst 8 WMMAs
        FragB16 a[2];
#pragma unroll
        for (int i = 0; i < 2; ++i) {
            int am = wm * 32 + i * 16 + ln;
            a[i].u[0] = *(const u32x4*)(&lA[cur][am * LDT + 16 * h]);
            a[i].u[1] = *(const u32x4*)(&lA[cur][am * LDT + 16 * h + 8]);
        }
        FragB16 b[4];
#pragma unroll
        for (int j = 0; j < 4; ++j) {
            int bn = wn * 64 + j * 16 + ln;
            b[j].u[0] = *(const u32x4*)(&lB[cur][bn * LDT + 16 * h]);
            b[j].u[1] = *(const u32x4*)(&lB[cur][bn * LDT + 16 * h + 8]);
        }
#pragma unroll
        for (int i = 0; i < 2; ++i)
#pragma unroll
            for (int j = 0; j < 4; ++j)
                acc[i][j] = __builtin_amdgcn_wmma_f32_16x16x32_bf16(
                    false, a[i].v, false, b[j].v, (short)0, acc[i][j], false, false);

        __syncthreads();   // single barrier per K-step (double-buffered)
    }

    // Epilogue: C-layout element (M = r+8h, N = 16j+ln)
#pragma unroll
    for (int i = 0; i < 2; ++i) {
#pragma unroll
        for (int j = 0; j < 4; ++j) {
#pragma unroll
            for (int r = 0; r < 8; ++r) {
                int mrow = rowBase + wm * 32 + i * 16 + r + 8 * h;
                int n    = colBase + wn * 64 + j * 16 + ln;
                float val = acc[i][j][r] + bias[n];
                unsigned short bv = f32_to_bf16(val);
                int which = n >> 10;          // 0=q 1=k 2=v
                int cc = n & 1023;
                int hh = cc >> 6;
                int dd = cc & 63;
                int bb = mrow >> 11;          // / T
                int t  = mrow & 2047;
                int bh = bb * 16 + hh;
                if (which == 0)      qb[((size_t)bh * T + t) * 64 + dd] = bv;
                else if (which == 1) kb[((size_t)bh * T + t) * 64 + dd] = bv;
                else                 vt[((size_t)bh * 64 + dd) * T + t] = bv;
            }
        }
    }
}

// ---------------------------------------------------------------------------
// Stage 2: causal flash attention (one wave per 16-query tile)
//   q,k: [BH,T,64] bf16 ; v transposed: [BH,64,T] bf16 ; out yb: [B,T,1024]
// ---------------------------------------------------------------------------
__global__ __launch_bounds__(256) void attn_flash_bf16(
    const unsigned short* __restrict__ qb,
    const unsigned short* __restrict__ kb,
    const unsigned short* __restrict__ vt,
    unsigned short* __restrict__ yb)
{
    const int T = 2048, D = 64;
    __shared__ __align__(16) unsigned short lP[8 * 16 * 40]; // per-wave 16x32 P

    const int tid  = threadIdx.x;
    const int wave = tid >> 5;
    const int lane = tid & 31;
    const int h    = lane >> 4;
    const int ln   = lane & 15;
    const int bh    = blockIdx.y;
    const int qBase = blockIdx.x * 128 + wave * 16;
    unsigned short* myP = &lP[wave * 16 * 40];

    const unsigned short* qptr = qb + (size_t)bh * T * D;
    const unsigned short* kptr = kb + (size_t)bh * T * D;
    const unsigned short* vptr = vt + (size_t)bh * D * T;

    // Q fragments: A-layout, two K-chunks covering d=[0,32) and [32,64)
    FragB16 aq[2];
    {
        const unsigned short* qrow = qptr + (size_t)(qBase + ln) * D;
#pragma unroll
        for (int c = 0; c < 2; ++c) {
            aq[c].u[0] = *(const u32x4*)(qrow + 32 * c + 16 * h);
            aq[c].u[1] = *(const u32x4*)(qrow + 32 * c + 16 * h + 8);
        }
    }

    v8f o[4];
#pragma unroll
    for (int j = 0; j < 4; ++j) o[j] = (v8f)0.0f;
    float mrow[8], lrow[8];
#pragma unroll
    for (int r = 0; r < 8; ++r) { mrow[r] = -__builtin_inff(); lrow[r] = 0.0f; }

    const float scale = 0.125f;  // 1/sqrt(64)
    const int nkb = (qBase + 16 + 31) / 32;   // 32-key blocks, causal bound

    for (int kblk = 0; kblk < nkb; ++kblk) {
        const int kBase = kblk * 32;

        // ---- S = Q K^T for two 16-key sub-tiles ----
        v8f s[2];
#pragma unroll
        for (int t = 0; t < 2; ++t) {
            const unsigned short* krow = kptr + (size_t)(kBase + 16 * t + ln) * D;
            FragB16 b0, b1;                     // B-layout: k = 16h+e (contig)
            b0.u[0] = *(const u32x4*)(krow + 16 * h);
            b0.u[1] = *(const u32x4*)(krow + 16 * h + 8);
            b1.u[0] = *(const u32x4*)(krow + 32 + 16 * h);
            b1.u[1] = *(const u32x4*)(krow + 32 + 16 * h + 8);
            v8f z = (v8f)0.0f;
            z = __builtin_amdgcn_wmma_f32_16x16x32_bf16(
                false, aq[0].v, false, b0.v, (short)0, z, false, false);
            z = __builtin_amdgcn_wmma_f32_16x16x32_bf16(
                false, aq[1].v, false, b1.v, (short)0, z, false, false);
            s[t] = z;
        }

        // ---- scale + causal mask ----
        const bool needMask = (kBase + 31 > qBase);
#pragma unroll
        for (int t = 0; t < 2; ++t) {
#pragma unroll
            for (int r = 0; r < 8; ++r) {
                float v = s[t][r] * scale;
                if (needMask) {
                    int q   = qBase + r + 8 * h;
                    int key = kBase + 16 * t + ln;
                    if (key > q) v = -__builtin_inff();
                }
                s[t][r] = v;
            }
        }

        // ---- online softmax (row reductions over 16-lane groups) ----
#pragma unroll
        for (int r = 0; r < 8; ++r) {
            float rm = fmaxf(s[0][r], s[1][r]);
#pragma unroll
            for (int off = 1; off < 16; off <<= 1)
                rm = fmaxf(rm, __shfl_xor(rm, off, 32));
            float mnew = fmaxf(mrow[r], rm);
            float corr = __expf(mrow[r] - mnew);
            float p0 = __expf(s[0][r] - mnew);
            float p1 = __expf(s[1][r] - mnew);
            float rs = p0 + p1;
#pragma unroll
            for (int off = 1; off < 16; off <<= 1)
                rs += __shfl_xor(rs, off, 32);
            lrow[r] = lrow[r] * corr + rs;
            mrow[r] = mnew;
#pragma unroll
            for (int j = 0; j < 4; ++j) o[j][r] *= corr;
            // stage P (C-layout -> LDS [m][key]), per-wave private patch
            int m = r + 8 * h;
            myP[m * 40 + ln]      = f32_to_bf16(p0);
            myP[m * 40 + 16 + ln] = f32_to_bf16(p1);
        }

        // ---- reload P in A-layout (K = 32 keys) ----
        FragB16 ap;
        ap.u[0] = *(const u32x4*)(&myP[ln * 40 + 16 * h]);
        ap.u[1] = *(const u32x4*)(&myP[ln * 40 + 16 * h + 8]);

        // ---- O += P V : V fragments contiguous from transposed V ----
#pragma unroll
        for (int j = 0; j < 4; ++j) {
            const unsigned short* vrow = vptr + (size_t)(16 * j + ln) * T + kBase;
            FragB16 bv;
            bv.u[0] = *(const u32x4*)(vrow + 16 * h);
            bv.u[1] = *(const u32x4*)(vrow + 16 * h + 8);
            o[j] = __builtin_amdgcn_wmma_f32_16x16x32_bf16(
                false, ap.v, false, bv.v, (short)0, o[j], false, false);
        }
    }

    // ---- normalize and write yb [B,T,1024] bf16 ----
    const int b  = bh >> 4;
    const int hh = bh & 15;
#pragma unroll
    for (int r = 0; r < 8; ++r) {
        float inv = 1.0f / lrow[r];
        int q = qBase + r + 8 * h;
        size_t base = ((size_t)(b * T + q)) * 1024 + hh * 64;
#pragma unroll
        for (int j = 0; j < 4; ++j)
            yb[base + 16 * j + ln] = f32_to_bf16(o[j][r] * inv);
    }
}

// ---------------------------------------------------------------------------
// Stage 3: output projection. A = yb [8192,1024] bf16, Bt = W_out^T bf16
// Epilogue: +b_out, write f32 to d_out [B,T,C]
// ---------------------------------------------------------------------------
__global__ __launch_bounds__(256) void gemm_out_bf16(
    const unsigned short* __restrict__ A,
    const unsigned short* __restrict__ Bt,   // [N,K] pre-transposed
    const float* __restrict__ bias,
    float* __restrict__ Out)
{
    const int K = 1024, N = 1024;
    __shared__ __align__(16) unsigned short lA[2][BM * LDT];
    __shared__ __align__(16) unsigned short lB[2][BN * LDT];

    const int tid  = threadIdx.x;
    const int wave = tid >> 5;
    const int lane = tid & 31;
    const int h    = lane >> 4;
    const int ln   = lane & 15;
    const int wm   = wave & 3;
    const int wn   = wave >> 2;
    const int rowBase = blockIdx.y * BM;
    const int colBase = blockIdx.x * BN;

    v8f acc[2][4];
#pragma unroll
    for (int i = 0; i < 2; ++i)
#pragma unroll
        for (int j = 0; j < 4; ++j) acc[i][j] = (v8f)0.0f;

    const int fr = tid >> 2;
    const int fc = (tid & 3) * 8;

#pragma unroll
    for (int it = 0; it < 2; ++it) {
        int r = fr + it * 64;
        *(u32x4*)(&lA[0][r * LDT + fc]) =
            *(const u32x4*)(A + (size_t)(rowBase + r) * K + fc);
        *(u32x4*)(&lB[0][r * LDT + fc]) =
            *(const u32x4*)(Bt + (size_t)(colBase + r) * K + fc);
    }
    __syncthreads();

    for (int k0 = 0; k0 < K; k0 += BK) {
        const int cur = (k0 >> 5) & 1;
        const int nxt = cur ^ 1;

        if (k0 + BK < K) {
            if (k0 + 2 * BK < K) {
                __builtin_prefetch(A  + (size_t)(rowBase + fr) * K + k0 + 2 * BK + fc, 0, 3);
                __builtin_prefetch(Bt + (size_t)(colBase + fr) * K + k0 + 2 * BK + fc, 0, 3);
            }
#pragma unroll
            for (int it = 0; it < 2; ++it) {
                int r = fr + it * 64;
                *(u32x4*)(&lA[nxt][r * LDT + fc]) =
                    *(const u32x4*)(A + (size_t)(rowBase + r) * K + k0 + BK + fc);
                *(u32x4*)(&lB[nxt][r * LDT + fc]) =
                    *(const u32x4*)(Bt + (size_t)(colBase + r) * K + k0 + BK + fc);
            }
        }

        FragB16 a[2];
#pragma unroll
        for (int i = 0; i < 2; ++i) {
            int am = wm * 32 + i * 16 + ln;
            a[i].u[0] = *(const u32x4*)(&lA[cur][am * LDT + 16 * h]);
            a[i].u[1] = *(const u32x4*)(&lA[cur][am * LDT + 16 * h + 8]);
        }
        FragB16 b[4];
#pragma unroll
        for (int j = 0; j < 4; ++j) {
            int bn = wn * 64 + j * 16 + ln;
            b[j].u[0] = *(const u32x4*)(&lB[cur][bn * LDT + 16 * h]);
            b[j].u[1] = *(const u32x4*)(&lB[cur][bn * LDT + 16 * h + 8]);
        }
#pragma unroll
        for (int i = 0; i < 2; ++i)
#pragma unroll
            for (int j = 0; j < 4; ++j)
                acc[i][j] = __builtin_amdgcn_wmma_f32_16x16x32_bf16(
                    false, a[i].v, false, b[j].v, (short)0, acc[i][j], false, false);

        __syncthreads();
    }

#pragma unroll
    for (int i = 0; i < 2; ++i) {
#pragma unroll
        for (int j = 0; j < 4; ++j) {
#pragma unroll
            for (int r = 0; r < 8; ++r) {
                int mrow = rowBase + wm * 32 + i * 16 + r + 8 * h;
                int n    = colBase + wn * 64 + j * 16 + ln;
                Out[(size_t)mrow * N + n] = acc[i][j][r] + bias[n];
            }
        }
    }
}

// ---------------------------------------------------------------------------
// Host-side launcher
// ---------------------------------------------------------------------------
extern "C" void kernel_launch(void* const* d_in, const int* in_sizes, int n_in,
                              void* d_out, int out_size, void* d_ws, size_t ws_size,
                              hipStream_t stream) {
    (void)in_sizes; (void)n_in; (void)out_size; (void)ws_size;

    const int Bsz = 4, T = 2048, C = 1024, H = 16, Dh = 64;
    const int M = Bsz * T;            // 8192
    const int BH = Bsz * H;           // 64

    const float* x     = (const float*)d_in[0];
    const float* W_in  = (const float*)d_in[1];
    const float* b_in  = (const float*)d_in[2];
    const float* W_out = (const float*)d_in[3];
    const float* b_out = (const float*)d_in[4];
    float* out = (float*)d_out;

    char* ws = (char*)d_ws;
    size_t off = 0;
    unsigned short* xb  = (unsigned short*)(ws + off); off += (size_t)M * C * 2;         // 16 MB
    unsigned short* wib = (unsigned short*)(ws + off); off += (size_t)C * 3 * C * 2;     //  6 MB (W_in^T)
    unsigned short* wob = (unsigned short*)(ws + off); off += (size_t)C * C * 2;         //  2 MB (W_out^T)
    unsigned short* qb  = (unsigned short*)(ws + off); off += (size_t)BH * T * Dh * 2;   // 16 MB
    unsigned short* kb  = (unsigned short*)(ws + off); off += (size_t)BH * T * Dh * 2;   // 16 MB
    unsigned short* vt  = (unsigned short*)(ws + off); off += (size_t)BH * Dh * T * 2;   // 16 MB
    unsigned short* yb  = (unsigned short*)(ws + off); off += (size_t)M * C * 2;         // 16 MB

    // Stage 0: convert activations; transpose-convert weights
    cvt_f32_to_bf16<<<4096, 256, 0, stream>>>(x, xb, M * C);
    cvt_transpose_f32_to_bf16<<<4096, 256, 0, stream>>>(W_in,  wib, C, 3 * C);
    cvt_transpose_f32_to_bf16<<<2048, 256, 0, stream>>>(W_out, wob, C, C);

    // Stage 1: QKV projection (+bias, scatter to attention layouts)
    gemm_qkv_bf16<<<dim3(3 * C / BN, M / BM), 256, 0, stream>>>(
        xb, wib, b_in, qb, kb, vt);

    // Stage 2: causal flash attention
    attn_flash_bf16<<<dim3(T / 128, BH), 256, 0, stream>>>(qb, kb, vt, yb);

    // Stage 3: output projection (+bias, f32 out)
    gemm_out_bf16<<<dim3(C / BN, M / BM), 256, 0, stream>>>(yb, wob, b_out, out);
}
